// MaskedLinearSharedWeights_62852551409792
// MI455X (gfx1250) — compile-verified, matrix operand order checked by reading, not run on hardware
//
#include <hip/hip_runtime.h>
#include <hip/hip_bf16.h>
#include <stdint.h>

// CDNA5 / gfx1250 wave32 WMMA types
typedef __attribute__((ext_vector_type(16))) __bf16 v16bf;
typedef __attribute__((ext_vector_type(8)))  float  v8f;

#define TILE_N 128
#define TILE_O 128
#define KC     32
#define LDR    20   // uint32 per LDS row = 40 bf16 (32 data + 8 pad) -> bank spread, 16B aligned

// round-to-nearest-even f32 -> bf16 bits
__device__ __forceinline__ uint32_t bf16_rne(float f) {
  uint32_t u = __float_as_uint(f);
  return (u + 0x7FFFu + ((u >> 16) & 1u)) >> 16;
}

union Frag { uint4 u4[2]; v16bf v; };

__global__ __launch_bounds__(256) void masked_linear_wmma(
    const float* __restrict__ x, const float* __restrict__ y,
    const float* __restrict__ w, const float* __restrict__ mask,
    const float* __restrict__ bias, float* __restrict__ out)
{
  constexpr int T = 1024, O = 1024, NN = 256, G = 8;

  __shared__ uint32_t Ah[TILE_N * LDR];
  __shared__ uint32_t Al[TILE_N * LDR];
  __shared__ uint32_t Wh[TILE_O * LDR];
  __shared__ uint32_t Wl[TILE_O * LDR];

  const int b   = blockIdx.z;
  const int n0  = blockIdx.y * TILE_N;
  const int o0  = blockIdx.x * TILE_O;
  const int tid = threadIdx.x;
  const int lane  = tid & 31;
  const int wave  = tid >> 5;
  const int waveN = wave & 1;   // which 64-row N slab
  const int waveO = wave >> 1;  // which 32-col O slab

  float yb[G];
  #pragma unroll
  for (int g = 0; g < G; ++g) yb[g] = y[b * G + g];

  // WMMA 16-bit A/B per-lane layout (ISA 7.12.2):
  // lanes 0-15: row = lane, K = 0..7 (frag[0..7]) and 16..23 (frag[8..15])
  // lanes16-31: row = lane-16, K = 8..15 and 24..31
  const int lrow = lane & 15;
  const int koff = (lane < 16) ? 0 : 8;  // bf16 offset in K-chunk
  const int kuo  = koff >> 1;            // uint32 offset

  v8f acc[4][2];
  #pragma unroll
  for (int mt = 0; mt < 4; ++mt)
    #pragma unroll
    for (int ot = 0; ot < 2; ++ot) acc[mt][ot] = {};

  for (int kc = 0; kc < T; kc += KC) {
    // ---- stage x tile (128 rows x 32 K, f32) as bf16 hi/lo into LDS ----
    #pragma unroll
    for (int i = 0; i < 4; ++i) {
      int idx = tid + i * 256;        // 0..1023 over a 128 x 8 float4 grid
      int r   = idx >> 3;
      int c4  = idx & 7;
      const float4 v = *(const float4*)(x + ((size_t)(b * NN + n0 + r) * T + kc + c4 * 4));
      uint32_t h0 = bf16_rne(v.x), h1 = bf16_rne(v.y);
      uint32_t h2 = bf16_rne(v.z), h3 = bf16_rne(v.w);
      uint32_t l0 = bf16_rne(v.x - __uint_as_float(h0 << 16));
      uint32_t l1 = bf16_rne(v.y - __uint_as_float(h1 << 16));
      uint32_t l2 = bf16_rne(v.z - __uint_as_float(h2 << 16));
      uint32_t l3 = bf16_rne(v.w - __uint_as_float(h3 << 16));
      Ah[r * LDR + c4 * 2    ] = h0 | (h1 << 16);
      Ah[r * LDR + c4 * 2 + 1] = h2 | (h3 << 16);
      Al[r * LDR + c4 * 2    ] = l0 | (l1 << 16);
      Al[r * LDR + c4 * 2 + 1] = l2 | (l3 << 16);
    }

    // ---- stage genre-mixed + masked weight tile (128 O-rows x 32 K) ----
    #pragma unroll
    for (int i = 0; i < 4; ++i) {
      int idx = tid + i * 256;
      int r   = idx >> 3;
      int c4  = idx & 7;
      size_t base = (size_t)(o0 + r) * T + kc + c4 * 4;
      const float4 mv = *(const float4*)(mask + base);
      float ax = 0.f, ay = 0.f, az = 0.f, aw = 0.f;
      #pragma unroll
      for (int g = 0; g < G; ++g) {
        const float4 wv = *(const float4*)(w + (size_t)g * O * T + base);
        ax = fmaf(yb[g], wv.x, ax);
        ay = fmaf(yb[g], wv.y, ay);
        az = fmaf(yb[g], wv.z, az);
        aw = fmaf(yb[g], wv.w, aw);
      }
      ax *= mv.x; ay *= mv.y; az *= mv.z; aw *= mv.w;
      uint32_t h0 = bf16_rne(ax), h1 = bf16_rne(ay);
      uint32_t h2 = bf16_rne(az), h3 = bf16_rne(aw);
      uint32_t l0 = bf16_rne(ax - __uint_as_float(h0 << 16));
      uint32_t l1 = bf16_rne(ay - __uint_as_float(h1 << 16));
      uint32_t l2 = bf16_rne(az - __uint_as_float(h2 << 16));
      uint32_t l3 = bf16_rne(aw - __uint_as_float(h3 << 16));
      Wh[r * LDR + c4 * 2    ] = h0 | (h1 << 16);
      Wh[r * LDR + c4 * 2 + 1] = h2 | (h3 << 16);
      Wl[r * LDR + c4 * 2    ] = l0 | (l1 << 16);
      Wl[r * LDR + c4 * 2 + 1] = l2 | (l3 << 16);
    }
    __syncthreads();

    // ---- WMMA compute: acc += Ah*Wh + Ah*Wl + Al*Wh (bf16x3 ~ f32) ----
    Frag bh[2], bl[2];
    #pragma unroll
    for (int ot = 0; ot < 2; ++ot) {
      int orow = waveO * 32 + ot * 16 + lrow;
      const uint4* ph = (const uint4*)&Wh[orow * LDR + kuo];
      bh[ot].u4[0] = ph[0]; bh[ot].u4[1] = ph[2];   // K+0..7, K+16..23 (or 8.., 24..)
      const uint4* pl = (const uint4*)&Wl[orow * LDR + kuo];
      bl[ot].u4[0] = pl[0]; bl[ot].u4[1] = pl[2];
    }
    #pragma unroll
    for (int mt = 0; mt < 4; ++mt) {
      int mrow = waveN * 64 + mt * 16 + lrow;
      Frag ah, al;
      const uint4* pah = (const uint4*)&Ah[mrow * LDR + kuo];
      ah.u4[0] = pah[0]; ah.u4[1] = pah[2];
      const uint4* pal = (const uint4*)&Al[mrow * LDR + kuo];
      al.u4[0] = pal[0]; al.u4[1] = pal[2];
      #pragma unroll
      for (int ot = 0; ot < 2; ++ot) {
        acc[mt][ot] = __builtin_amdgcn_wmma_f32_16x16x32_bf16(
            false, ah.v, false, bh[ot].v, (short)0, acc[mt][ot], false, false);
        acc[mt][ot] = __builtin_amdgcn_wmma_f32_16x16x32_bf16(
            false, ah.v, false, bl[ot].v, (short)0, acc[mt][ot], false, false);
        acc[mt][ot] = __builtin_amdgcn_wmma_f32_16x16x32_bf16(
            false, al.v, false, bh[ot].v, (short)0, acc[mt][ot], false, false);
      }
    }
    __syncthreads();
  }

  // ---- epilogue: C/D layout -> lane = N col (o), VGPR e + lane-half = M row ----
  const int mhalf = (lane < 16) ? 0 : 8;
  #pragma unroll
  for (int ot = 0; ot < 2; ++ot) {
    int o = o0 + waveO * 32 + ot * 16 + lrow;
    float bv = bias[o];
    #pragma unroll
    for (int mt = 0; mt < 4; ++mt) {
      int mbase = n0 + waveN * 64 + mt * 16 + mhalf;
      #pragma unroll
      for (int e = 0; e < 8; ++e) {
        out[((size_t)b * NN + mbase + e) * O + o] = acc[mt][ot][e] + bv;
      }
    }
  }
}

extern "C" void kernel_launch(void* const* d_in, const int* in_sizes, int n_in,
                              void* d_out, int out_size, void* d_ws, size_t ws_size,
                              hipStream_t stream) {
  const float* x    = (const float*)d_in[0];  // [B,N,T]
  const float* y    = (const float*)d_in[1];  // [B,G]
  const float* w    = (const float*)d_in[2];  // [G,O,T]
  const float* mask = (const float*)d_in[3];  // [O,T]
  const float* bias = (const float*)d_in[4];  // [O]
  float* out = (float*)d_out;                 // [B,N,O]

  const int B = in_sizes[1] / 8;              // y is [B, 8]
  dim3 grid(1024 / TILE_O, 256 / TILE_N, B);  // (8, 2, 32)
  masked_linear_wmma<<<grid, 256, 0, stream>>>(x, y, w, mask, bias, out);
}
